// TernaryMLP_31868657336981
// MI455X (gfx1250) — matrix operand order repaired and checked by reading, not compiled, not run on hardware
//
#include <hip/hip_runtime.h>
#include <hip/hip_bf16.h>

typedef _Float16 f16;
typedef __attribute__((ext_vector_type(16))) _Float16 v16h;
typedef __attribute__((ext_vector_type(8)))  _Float16 v8h;
typedef __attribute__((ext_vector_type(8)))  float    v8f;
typedef __attribute__((ext_vector_type(4)))  unsigned int u32x4;
typedef __attribute__((ext_vector_type(8)))  int      i32x8;
typedef __attribute__((ext_vector_type(4)))  int      i32x4;

#define BM  128   // block rows (batch)
#define BNT 128   // block cols (features)
#define BK  32    // K step = WMMA K
#define LDA 40    // LDS row stride (f16) for A tile  -> conflict-free
#define LDW 40    // LDS row stride (f16) for W tile  -> conflict-free (TDM pad)

#ifndef __has_builtin
#define __has_builtin(x) 0
#endif
#define HAS_TDM (__has_builtin(__builtin_amdgcn_tensor_load_to_lds) && \
                 __has_builtin(__builtin_amdgcn_s_wait_tensorcnt))

// ---------------- mean(|w|) partial reduction ----------------
__global__ __launch_bounds__(256)
void absum_kernel(const float* __restrict__ w, long n, float* __restrict__ out) {
    __shared__ float red[256];
    float s = 0.f;
    for (long i = blockIdx.x * 256l + threadIdx.x; i < n; i += 256l * gridDim.x)
        s += fabsf(w[i]);
    red[threadIdx.x] = s;
    __syncthreads();
    for (int off = 128; off > 0; off >>= 1) {
        if ((int)threadIdx.x < off) red[threadIdx.x] += red[threadIdx.x + off];
        __syncthreads();
    }
    if (threadIdx.x == 0) atomicAdd(out, red[0]);
}

// ------- ternary quantize to f16 {-1,0,1} into padded (Npad x Kp) buffer -------
__global__ __launch_bounds__(256)
void quant_kernel(const float* __restrict__ w, f16* __restrict__ q,
                  const float* __restrict__ absum, float rcnt,
                  int N, int K, int Kp) {
    long idx = blockIdx.x * 256l + threadIdx.x;
    int n = (int)(idx / Kp);
    int k = (int)(idx % Kp);
    f16 v = (f16)0.f;
    if (n < N && k < K) {
        float alpha = absum[0] * rcnt + 1e-8f;
        float t = rintf(w[(long)n * K + k] / alpha);
        v = (f16)fminf(1.f, fmaxf(-1.f, t));
    }
    q[idx] = v;
}

#if HAS_TDM
// Issue one TDM 2-D tile load: 32(K, contiguous) x 128(N rows) f16 panel into LDS,
// LDS-padded 4 dwords per 16-dword row => 40-half row stride.
__device__ __forceinline__ void tdm_load_w(unsigned long long gaddr, unsigned ldsOff, int Kp) {
    u32x4 g0 = { 1u,                                  // count=1, user mode, no gather
                 ldsOff,                              // D#.lds_addr (bytes)
                 (unsigned)gaddr,                     // global_addr[31:0]
                 (unsigned)((gaddr >> 32) & 0x1FFFFFFu) | (2u << 30) }; // addr[56:32] | type=2
    unsigned td0 = (unsigned)Kp;                      // tensor_dim0 (elements)
    unsigned td1 = (unsigned)BNT;                     // tensor_dim1 (rows)
    i32x8 g1 = { (int)((1u << 16)                     // data_size = 2 bytes
                     | (1u << 20)                     // pad_enable
                     | (3u << 22)                     // pad_interval: 16 dwords (one 32-f16 row)
                     | (3u << 25)),                   // pad_amount: 4 dwords (8 f16)
                 (int)((td0 & 0xFFFFu) << 16),                        // tensor_dim0[15:0]
                 (int)(((td0 >> 16) & 0xFFFFu) | ((td1 & 0xFFFFu) << 16)), // td0 hi | td1 lo
                 (int)(((td1 >> 16) & 0xFFFFu) | (32u << 16)),        // td1 hi | tile_dim0=32
                 (int)BNT,                                            // tile_dim1=128, tile_dim2=0
                 (int)td0,                                            // tensor_dim0_stride = Kp
                 0, 0 };
    i32x4 z4 = {0, 0, 0, 0};
#if __clang_major__ >= 23
    i32x8 z8 = {0, 0, 0, 0, 0, 0, 0, 0};
    __builtin_amdgcn_tensor_load_to_lds(g0, g1, z4, z4, z8, 0);
#else
    __builtin_amdgcn_tensor_load_to_lds(g0, g1, z4, z4, 0);
#endif
}
#endif

// ---------------- WMMA GEMM: C[B x Nact] = A[B x Ka] * Wq^T, scaled by alpha ----------------
__global__ __launch_bounds__(256)
void gemm_kernel(const float* __restrict__ A, const f16* __restrict__ Wq,
                 float* __restrict__ C,
                 const float* __restrict__ absum, float rcnt,
                 int Nact, int Kp, int Ka,
                 float* __restrict__ gsum, float* __restrict__ gsq, int doStats) {
    __shared__ f16   As[BM * LDA];
    __shared__ f16   Ws[2 * BNT * LDW];     // double-buffered TDM destination
    __shared__ float lds_s[BNT];
    __shared__ float lds_s2[BNT];

    const int tid  = threadIdx.x;
    const int lane = tid & 31;              // wave32
    const int wid  = tid >> 5;              // 8 waves
    const int wm   = wid & 3;               // 4 waves along M (32 rows each)
    const int wn   = wid >> 2;              // 2 waves along N (64 cols each)
    const int m0   = blockIdx.y * BM;
    const int n0   = blockIdx.x * BNT;

    const float alpha = absum[0] * rcnt + 1e-8f;

    if (tid < BNT) { lds_s[tid] = 0.f; lds_s2[tid] = 0.f; }

    v8f acc[2][4] = {};

    // A staging: 2 threads per row, 16 f32 each
    const int arow = tid >> 1;
    const int acol = (tid & 1) * 16;
    const float* Arow = A + (long)(m0 + arow) * Ka;
    f16* dstA = &As[arow * LDA + acol];

    const int nl    = lane & 15;
    const int abase = wm * 32 + nl;         // A-frag LDS row base
    const int khalf = (lane >> 4) * 8;      // A frag: K 0-7/16-23 vs 8-15/24-31
    const int kbcol = (lane >> 4) * 8;      // B frag: K 0-7/16-23 vs 8-15/24-31

    const f16* WqBlk = Wq + (long)n0 * Kp;

#if HAS_TDM
    const unsigned long long wgbase = (unsigned long long)(unsigned long long)WqBlk;
    const unsigned wsOff = (unsigned)(unsigned long long)(&Ws[0]);  // LDS byte offset
    if (wid == 0) tdm_load_w(wgbase, wsOff, Kp);     // prime the pipeline (buf 0)
#endif

    for (int kt = 0; kt < Kp; kt += BK) {
        const int buf = (kt >> 5) & 1;
        // ---- stage A tile (f32 -> f16) into LDS ----
        const int kc = kt + acol;
        if (kc < Ka) {   // Ka multiple of 16 => whole chunk in range
            const float4* ap = (const float4*)(Arow + kc);
            float4 f0 = ap[0], f1 = ap[1], f2 = ap[2], f3 = ap[3];
            v8h h0 = {(f16)f0.x,(f16)f0.y,(f16)f0.z,(f16)f0.w,
                      (f16)f1.x,(f16)f1.y,(f16)f1.z,(f16)f1.w};
            v8h h1 = {(f16)f2.x,(f16)f2.y,(f16)f2.z,(f16)f2.w,
                      (f16)f3.x,(f16)f3.y,(f16)f3.z,(f16)f3.w};
            *(v8h*)dstA       = h0;
            *(v8h*)(dstA + 8) = h1;
            if (kc + BK < Ka)                        // global_prefetch_b8 next tile
                __builtin_prefetch((const void*)(Arow + kc + BK), 0, 3);
        } else {
            v8h z = {};
            *(v8h*)dstA       = z;
            *(v8h*)(dstA + 8) = z;
        }
#if HAS_TDM
        if (wid == 0) __builtin_amdgcn_s_wait_tensorcnt(0);  // current W buf ready
#else
        {   // fallback: cooperative W tile load, same padded layout
            const int wrow = tid >> 1;
            const int wc   = (tid & 1) * 16;
            v16h wv = *(const v16h*)(WqBlk + (long)wrow * Kp + kt + wc);
            union { v16h v; v8h h[2]; } u; u.v = wv;
            f16* dw = &Ws[buf * (BNT * LDW) + wrow * LDW + wc];
            *(v8h*)dw       = u.h[0];
            *(v8h*)(dw + 8) = u.h[1];
        }
#endif
        __syncthreads();
#if HAS_TDM
        if (wid == 0 && kt + BK < Kp)                // overlap next DMA with compute
            tdm_load_w(wgbase + (unsigned long long)(kt + BK) * 2ull,
                       wsOff + (unsigned)((buf ^ 1) * (BNT * LDW * 2)), Kp);
#endif
        // ---- A fragments (per ISA 16-bit A layout) ----
        union { v16h v; v8h h[2]; } afr[2];
#pragma unroll
        for (int mi = 0; mi < 2; ++mi) {
            const f16* p = &As[(abase + mi * 16) * LDA + khalf];
            afr[mi].h[0] = *(const v8h*)p;
            afr[mi].h[1] = *(const v8h*)(p + 16);
        }
        // ---- B fragments: K 0-7/16-23 vs 8-15/24-31 of column N=lane&15, from LDS panel ----
        const f16* wsb = &Ws[buf * (BNT * LDW)];
#pragma unroll
        for (int nj = 0; nj < 4; ++nj) {
            const f16* p = &wsb[(wn * 64 + nj * 16 + nl) * LDW + kbcol];
            union { v16h v; v8h h[2]; } bfr;
            bfr.h[0] = *(const v8h*)p;
            bfr.h[1] = *(const v8h*)(p + 16);
#pragma unroll
            for (int mi = 0; mi < 2; ++mi)
                acc[mi][nj] = __builtin_amdgcn_wmma_f32_16x16x32_f16(
                    false, afr[mi].v, false, bfr.v,
                    (short)0, acc[mi][nj], false, false);
        }
        __syncthreads();
    }

    // ---- epilogue: scale by alpha, store, accumulate column stats ----
    const int mh = (lane >> 4) * 8;   // C layout: lanes 16-31 carry rows M+8..M+15
#pragma unroll
    for (int mi = 0; mi < 2; ++mi)
#pragma unroll
        for (int nj = 0; nj < 4; ++nj) {
            const int nloc = wn * 64 + nj * 16 + nl;
            const int ngl  = n0 + nloc;
            const int mg   = m0 + wm * 32 + mi * 16 + mh;
            const bool nok = ngl < Nact;
            float s = 0.f, s2 = 0.f;
#pragma unroll
            for (int r = 0; r < 8; ++r) {
                float v = acc[mi][nj][r] * alpha;
                if (nok) C[(long)(mg + r) * Nact + ngl] = v;
                s += v; s2 += v * v;
            }
            if (doStats) {                 // ds_add_f32
                atomicAdd(&lds_s[nloc],  s);
                atomicAdd(&lds_s2[nloc], s2);
            }
        }
    if (doStats) {
        __syncthreads();
        if (tid < BNT) {
            atomicAdd(&gsum[n0 + tid], lds_s[tid]);
            atomicAdd(&gsq [n0 + tid], lds_s2[tid]);
        }
    }
}

// ---------------- BatchNorm (training stats) + ReLU, in place ----------------
__global__ __launch_bounds__(256)
void bnrelu_kernel(float* __restrict__ y, const float* __restrict__ gsum,
                   const float* __restrict__ gsq,
                   const float* __restrict__ gamma, const float* __restrict__ beta,
                   int N, float invB) {
    long idx = blockIdx.x * 256l + threadIdx.x;     // float4 index
    int col = (int)(idx % (N >> 2)) * 4;
    float4 v = ((const float4*)y)[idx];
    float o[4] = {v.x, v.y, v.z, v.w};
#pragma unroll
    for (int j = 0; j < 4; ++j) {
        int n = col + j;
        float mean = gsum[n] * invB;
        float var  = gsq[n] * invB - mean * mean;
        float t = (o[j] - mean) * rsqrtf(var + 1e-5f) * gamma[n] + beta[n];
        o[j] = fmaxf(t, 0.f);
    }
    ((float4*)y)[idx] = make_float4(o[0], o[1], o[2], o[3]);
}

extern "C" void kernel_launch(void* const* d_in, const int* in_sizes, int n_in,
                              void* d_out, int out_size, void* d_ws, size_t ws_size,
                              hipStream_t stream) {
    (void)in_sizes; (void)n_in; (void)out_size; (void)ws_size;
    const float* x = (const float*)d_in[0];
    const float* w[4] = {(const float*)d_in[1], (const float*)d_in[2],
                         (const float*)d_in[3], (const float*)d_in[4]};
    const float* g[3] = {(const float*)d_in[5], (const float*)d_in[7], (const float*)d_in[9]};
    const float* b[3] = {(const float*)d_in[6], (const float*)d_in[8], (const float*)d_in[10]};
    float* out = (float*)d_out;

    char* ws = (char*)d_ws;
    size_t off = 0;
    auto take = [&](size_t bytes) -> char* {
        char* p = ws + off;
        off += (bytes + 255) & ~(size_t)255;
        return p;
    };
    float* y1 = (float*)take((size_t)32768 * 1024 * 4);
    float* y2 = (float*)take((size_t)32768 * 512 * 4);
    float* y3 = (float*)take((size_t)32768 * 256 * 4);
    f16* q[4];
    q[0] = (f16*)take((size_t)1024 * 800 * 2);
    q[1] = (f16*)take((size_t)512 * 1024 * 2);
    q[2] = (f16*)take((size_t)256 * 512 * 2);
    q[3] = (f16*)take((size_t)128 * 256 * 2);
    float* stats = (float*)take((size_t)(4 + 1792 * 2) * 4);
    float* absum = stats;            // 4
    float* gsum  = stats + 4;        // 1792 (1024 | 512 | 256)
    float* gsq   = stats + 4 + 1792; // 1792

    hipMemsetAsync(stats, 0, (4 + 1792 * 2) * sizeof(float), stream);

    const long wcnt[4] = {1024l * 784, 512l * 1024, 256l * 512, 10l * 256};
    const int  Nw[4]  = {1024, 512, 256, 10};
    const int  Kw[4]  = {784, 1024, 512, 256};
    const int  Npq[4] = {1024, 512, 256, 128};
    const int  Kpq[4] = {800, 1024, 512, 256};

    for (int i = 0; i < 4; ++i)
        absum_kernel<<<256, 256, 0, stream>>>(w[i], wcnt[i], absum + i);
    for (int i = 0; i < 4; ++i) {
        long total = (long)Npq[i] * Kpq[i];
        quant_kernel<<<(int)(total / 256), 256, 0, stream>>>(
            w[i], q[i], absum + i, 1.f / (float)wcnt[i], Nw[i], Kw[i], Kpq[i]);
    }

    const float invB = 1.f / 32768.f;
    // Layer 1: 784 -> 1024
    gemm_kernel<<<dim3(1024 / BNT, 32768 / BM), 256, 0, stream>>>(
        x, q[0], y1, absum + 0, 1.f / (float)wcnt[0], 1024, 800, 784, gsum, gsq, 1);
    bnrelu_kernel<<<32768 * 1024 / 4 / 256, 256, 0, stream>>>(
        y1, gsum, gsq, g[0], b[0], 1024, invB);
    // Layer 2: 1024 -> 512
    gemm_kernel<<<dim3(512 / BNT, 32768 / BM), 256, 0, stream>>>(
        y1, q[1], y2, absum + 1, 1.f / (float)wcnt[1], 512, 1024, 1024,
        gsum + 1024, gsq + 1024, 1);
    bnrelu_kernel<<<32768 * 512 / 4 / 256, 256, 0, stream>>>(
        y2, gsum + 1024, gsq + 1024, g[1], b[1], 512, invB);
    // Layer 3: 512 -> 256
    gemm_kernel<<<dim3(256 / BNT, 32768 / BM), 256, 0, stream>>>(
        y2, q[2], y3, absum + 2, 1.f / (float)wcnt[2], 256, 512, 512,
        gsum + 1536, gsq + 1536, 1);
    bnrelu_kernel<<<32768 * 256 / 4 / 256, 256, 0, stream>>>(
        y3, gsum + 1536, gsq + 1536, g[2], b[2], 256, invB);
    // Layer 4: 256 -> 10 (no BN/ReLU), direct to output
    gemm_kernel<<<dim3(1, 32768 / BM), 256, 0, stream>>>(
        y3, q[3], out, absum + 3, 1.f / (float)wcnt[3], 10, 256, 256,
        nullptr, nullptr, 0);
}
